// DIIN_62070867362582
// MI455X (gfx1250) — matrix operand order repaired and checked by reading, not compiled, not run on hardware
//
#include <hip/hip_runtime.h>
#include <hip/hip_bf16.h>

typedef __attribute__((ext_vector_type(16))) _Float16 v16h;
typedef __attribute__((ext_vector_type(8)))  float    v8f;
typedef __attribute__((ext_vector_type(4)))  float    f32x4;

__device__ __forceinline__ v8f wmma_f16(v16h a, v16h b, v8f c) {
  // v_wmma_f32_16x16x32_f16: (neg_a, A, neg_b, B, c_mod, C, reuse_a, reuse_b)
  return __builtin_amdgcn_wmma_f32_16x16x32_f16(false, a, false, b, (short)0, c, false, false);
}

// Pack four f32x4 chunks into a v16h fragment (f32 -> f16)
__device__ __forceinline__ v16h pack16(f32x4 q0, f32x4 q1, f32x4 q2, f32x4 q3) {
  v16h r;
#pragma unroll
  for (int e = 0; e < 4; ++e) {
    r[e]      = (_Float16)q0[e];
    r[e + 4]  = (_Float16)q1[e];
    r[e + 8]  = (_Float16)q2[e];
    r[e + 12] = (_Float16)q3[e];
  }
  return r;
}

// A fragment (16x32 f16): per-lane K runs are {k0+half*8 .. +7} and {+16..+23}
__device__ __forceinline__ v16h load_a_contig(const float* __restrict__ row, int k0, int half) {
  const f32x4* p = (const f32x4*)(row + k0 + half * 8);
  return pack16(p[0], p[1], p[4], p[5]);
}
// B fragment (32x16 f16): per-lane K runs are {k0+half*8 .. +7} and {+16..+23}
__device__ __forceinline__ v16h load_b_contig(const float* __restrict__ row, int k0, int half) {
  const f32x4* p = (const f32x4*)(row + k0 + half * 8);
  return pack16(p[0], p[1], p[4], p[5]);
}

// ---------------------------------------------------------------------------
// 1) translate: x = emb @ Wt^T + bt   (M=1024 rows, N=128, K=768), one launch
//    each for p and h (no per-lane pointer selects).
// ---------------------------------------------------------------------------
__global__ void k_translate(const float* __restrict__ emb, const float* __restrict__ Wt,
                            const float* __restrict__ bt, float* __restrict__ xout) {
  const int lane = threadIdx.x & 31, wave = threadIdx.x >> 5;
  const int half = lane >> 4, idx = lane & 15;
  const int m0 = blockIdx.x * 16;      // 64 m-tiles over 1024 rows
  const int n0 = wave * 16;            // 8 n-tiles over 128
  const float* src  = emb + (size_t)(m0 + idx) * 768;
  const float* wrow = Wt  + (size_t)(n0 + idx) * 768;
  v8f acc = {};
  for (int k0 = 0; k0 < 768; k0 += 32) {
    v16h a = load_a_contig(src, k0, half);
    v16h b = load_b_contig(wrow, k0, half);
    acc = wmma_f16(a, b, acc);
  }
  const int n = n0 + idx;
  const float bias = bt[n];
#pragma unroll
  for (int r = 0; r < 8; ++r) {
    int m = m0 + half * 8 + r;
    xout[(size_t)m * 128 + n] = acc[r] + bias;
  }
}

// ---------------------------------------------------------------------------
// 2) per-row dots: u = x.w_up, m = x.w_mid  (1024 rows x 128)
// ---------------------------------------------------------------------------
__global__ void k_rowdot(const float* __restrict__ x, const float* __restrict__ w_attn,
                         float* __restrict__ u, float* __restrict__ mv) {
  int r = blockIdx.x * blockDim.x + threadIdx.x;
  if (r >= 1024) return;
  const float* row = x + (size_t)r * 128;
  const float* wu = w_attn;
  const float* wm = w_attn + 128;
  float su = 0.f, sm = 0.f;
  for (int d = 0; d < 128; ++d) { float v = row[d]; su += v * wu[d]; sm += v * wm[d]; }
  u[r] = su; mv[r] = sm;
}

// ---------------------------------------------------------------------------
// 3) attention scores: S[b,i,j] = u[b,j] + m[b,i] + sum_d x[i,d]*wp[d]*x[j,d]
// ---------------------------------------------------------------------------
__global__ void k_scores(const float* __restrict__ x, const float* __restrict__ w_attn,
                         const float* __restrict__ u, const float* __restrict__ mv,
                         float* __restrict__ S) {
  const int lane = threadIdx.x & 31, wave = threadIdx.x >> 5;
  const int half = lane >> 4, idx = lane & 15;
  const int b = blockIdx.y, m0 = blockIdx.x * 16, n0 = wave * 16;
  const float* xb = x + (size_t)b * 128 * 128;
  const float* wp = w_attn + 256;
  const float* ar = xb + (size_t)(m0 + idx) * 128;
  const float* br = xb + (size_t)(n0 + idx) * 128;
  v8f acc = {};
  for (int k0 = 0; k0 < 128; k0 += 32) {
    const f32x4* pa = (const f32x4*)(ar + k0 + half * 8);
    const f32x4* pw = (const f32x4*)(wp + k0 + half * 8);
    v16h a = pack16(pa[0] * pw[0], pa[1] * pw[1], pa[4] * pw[4], pa[5] * pw[5]);
    v16h bb = load_b_contig(br, k0, half);
    acc = wmma_f16(a, bb, acc);
  }
  const int n = n0 + idx;
  const float uj = u[b * 128 + n];
#pragma unroll
  for (int r = 0; r < 8; ++r) {
    int m = m0 + half * 8 + r;
    S[((size_t)b * 128 + m) * 128 + n] = acc[r] + uj + mv[b * 128 + m];
  }
}

// ---------------------------------------------------------------------------
// 4) row softmax over j (row length 128)
// ---------------------------------------------------------------------------
__global__ void k_softmax(float* __restrict__ S) {
  __shared__ float red[128];
  const int r = blockIdx.x, t = threadIdx.x;
  float v = S[(size_t)r * 128 + t];
  red[t] = v; __syncthreads();
  for (int s = 64; s > 0; s >>= 1) { if (t < s) red[t] = fmaxf(red[t], red[t + s]); __syncthreads(); }
  const float mx = red[0]; __syncthreads();
  const float e = __expf(v - mx);
  red[t] = e; __syncthreads();
  for (int s = 64; s > 0; s >>= 1) { if (t < s) red[t] += red[t + s]; __syncthreads(); }
  S[(size_t)r * 128 + t] = e / red[0];
}

// ---------------------------------------------------------------------------
// 5) apply attention: out[b,i,d] = sum_j S[b,i,j] * x[b,j,d]
// ---------------------------------------------------------------------------
__global__ void k_attend(const float* __restrict__ S, const float* __restrict__ x,
                         float* __restrict__ out) {
  const int lane = threadIdx.x & 31, wave = threadIdx.x >> 5;
  const int half = lane >> 4, idx = lane & 15;
  const int b = blockIdx.y, m0 = blockIdx.x * 16, n0 = wave * 16;
  const float* Sb = S + (size_t)b * 128 * 128;
  const float* xb = x + (size_t)b * 128 * 128;
  const float* ar = Sb + (size_t)(m0 + idx) * 128;
  const int n = n0 + idx;
  v8f acc = {};
  for (int k0 = 0; k0 < 128; k0 += 32) {
    v16h a = load_a_contig(ar, k0, half);
    v16h bb;
#pragma unroll
    for (int e = 0; e < 16; ++e) bb[e] = (_Float16)xb[(size_t)(k0 + ((e < 8) ? e : e + 8) + half * 8) * 128 + n];
    acc = wmma_f16(a, bb, acc);
  }
#pragma unroll
  for (int r = 0; r < 8; ++r) {
    int m = m0 + half * 8 + r;
    out[((size_t)b * 128 + m) * 128 + n] = acc[r];
  }
}

// ---------------------------------------------------------------------------
// 6) interaction + 1x1 conv fused:
//    act[b,c,p,h] = bc[c] + sum_d Wc[c,d]*pe[b,p,d]*he[b,h,d]    (c < 38)
// ---------------------------------------------------------------------------
__global__ void k_interact(const float* __restrict__ peA, const float* __restrict__ heA,
                           const float* __restrict__ Wc, const float* __restrict__ bc,
                           float* __restrict__ act, int Cmax) {
  const int lane = threadIdx.x & 31, wave = threadIdx.x >> 5; // wave = m-tile (0..2)
  const int half = lane >> 4, idx = lane & 15;
  const int b = blockIdx.z, p = blockIdx.y, n0 = blockIdx.x * 16;
  const int m0 = wave * 16;
  const int m = m0 + idx;
  const float* pr = peA + ((size_t)b * 128 + p) * 128;
  const float* hr = heA + ((size_t)b * 128 + (n0 + idx)) * 128;
  const float* wr = Wc + (size_t)(m < 38 ? m : 0) * 128;   // clamp; masked below
  const bool mvalid = (m < 38);
  v8f acc = {};
  for (int k0 = 0; k0 < 128; k0 += 32) {
    const f32x4* pa = (const f32x4*)(wr + k0 + half * 8);
    const f32x4* pp = (const f32x4*)(pr + k0 + half * 8);
    f32x4 z = {};
    f32x4 q0 = mvalid ? pa[0] * pp[0] : z;
    f32x4 q1 = mvalid ? pa[1] * pp[1] : z;
    f32x4 q2 = mvalid ? pa[4] * pp[4] : z;
    f32x4 q3 = mvalid ? pa[5] * pp[5] : z;
    v16h a = pack16(q0, q1, q2, q3);
    v16h bb = load_b_contig(hr, k0, half);
    acc = wmma_f16(a, bb, acc);
  }
  const int n = n0 + idx;
#pragma unroll
  for (int r = 0; r < 8; ++r) {
    int c = m0 + half * 8 + r;
    if (c < 38)
      act[(((size_t)b * Cmax + c) * 128 + p) * 128 + n] = acc[r] + bc[c];
  }
}

// ---------------------------------------------------------------------------
// 7) dense layer: 3x3 conv (implicit GEMM, K = Cin*9), ReLU, concat in place.
//    One wave computes one 16-wide spatial tile for BOTH output-channel tiles
//    (Cout = 20 padded to 32), reusing one B fragment per K step.
//    k -> (c, ty, tx) decoded with an incremental odometer (no divisions in
//    the unrolled element loop).
// ---------------------------------------------------------------------------
__global__ void k_conv3(float* __restrict__ act, const float* __restrict__ w,
                        int Cmax, int Cin, int H, int W) {
  const int lane = threadIdx.x & 31, wave = threadIdx.x >> 5;
  const int half = lane >> 4, idx = lane & 15;
  const int nt = blockIdx.x * 8 + wave;
  const int b = blockIdx.y;
  const int HW = H * W;
  const int n = nt * 16 + idx;
  const int y = n / W, x0 = n % W;
  const float* in = act + (size_t)b * Cmax * HW;
  const float* w0 = w + (size_t)idx * (Cin * 9);
  const float* w1 = w + (size_t)(idx + 16 < 20 ? idx + 16 : 0) * (Cin * 9);
  const bool m1valid = (idx + 16 < 20);
  const int K = Cin * 9;
  v8f acc0 = {}, acc1 = {};
  for (int k0 = 0; k0 < K; k0 += 32) {
    v16h a0, a1, bb;
#pragma unroll
    for (int e = 0; e < 16; ++e) {
      int k = k0 + half * 16 + e;
      bool ok = (k < K);
      a0[e] = ok ? (_Float16)w0[k] : (_Float16)0.f;
      a1[e] = (ok && m1valid) ? (_Float16)w1[k] : (_Float16)0.f;
    }
    {
      const int kbase = k0 + half * 16;
      int c = kbase / 9;
      int rem = kbase - c * 9;
      int ty = rem / 3, tx = rem - ty * 3;     // one divide pair per K-step
#pragma unroll
      for (int e = 0; e < 16; ++e) {
        int k = kbase + e;
        _Float16 v = (_Float16)0.f;
        if (k < K) {
          int yy = y + ty - 1, xx = x0 + tx - 1;
          if (yy >= 0 && yy < H && xx >= 0 && xx < W)
            v = (_Float16)in[(size_t)c * HW + yy * W + xx];
        }
        bb[e] = v;
        if (++tx == 3) { tx = 0; if (++ty == 3) { ty = 0; ++c; } }
      }
    }
    acc0 = wmma_f16(a0, bb, acc0);
    acc1 = wmma_f16(a1, bb, acc1);
  }
  float* outb = act + ((size_t)b * Cmax + Cin) * HW;
#pragma unroll
  for (int r = 0; r < 8; ++r) {
    int m = half * 8 + r;
    float v0 = acc0[r];
    outb[(size_t)m * HW + n] = v0 > 0.f ? v0 : 0.f;
    int m1 = m + 16;
    if (m1 < 20) { float v1 = acc1[r]; outb[(size_t)m1 * HW + n] = v1 > 0.f ? v1 : 0.f; }
  }
}

// ---------------------------------------------------------------------------
// 8) transition 1x1 conv (no bias, no relu): out (B,Cout,H,W) contiguous
// ---------------------------------------------------------------------------
__global__ void k_conv1(const float* __restrict__ act, int CmaxIn, int Cin,
                        const float* __restrict__ w, float* __restrict__ out,
                        int Cout, int H, int W, int mtiles, int ntiles) {
  const int lane = threadIdx.x & 31, wave = threadIdx.x >> 5;
  const int half = lane >> 4, idx = lane & 15;
  const int id = blockIdx.x * 8 + wave;
  if (id >= mtiles * ntiles) return;      // wave-uniform: EXEC stays full
  const int m0 = (id / ntiles) * 16, n0 = (id % ntiles) * 16;
  const int b = blockIdx.y;
  const int HW = H * W;
  const float* in = act + (size_t)b * CmaxIn * HW;
  const int m = m0 + idx, n = n0 + idx;
  const float* wr = w + (size_t)(m < Cout ? m : 0) * Cin;
  const bool mvalid = (m < Cout);
  v8f acc = {};
  for (int k0 = 0; k0 < Cin; k0 += 32) {
    v16h a, bb;
#pragma unroll
    for (int e = 0; e < 16; ++e) {
      int k = k0 + half * 16 + e;
      a[e] = (mvalid && k < Cin) ? (_Float16)wr[k] : (_Float16)0.f;
    }
#pragma unroll
    for (int e = 0; e < 16; ++e) {
      int k = k0 + half * 16 + e;
      bb[e] = (k < Cin) ? (_Float16)in[(size_t)k * HW + n] : (_Float16)0.f;
    }
    acc = wmma_f16(a, bb, acc);
  }
#pragma unroll
  for (int r = 0; r < 8; ++r) {
    int mm = m0 + half * 8 + r;
    if (mm < Cout) out[((size_t)b * Cout + mm) * HW + n] = acc[r];
  }
}

// ---------------------------------------------------------------------------
// 9) 2x2 maxpool, writing into a buffer with channel capacity CmaxOut
// ---------------------------------------------------------------------------
__global__ void k_pool(const float* __restrict__ in, float* __restrict__ out,
                       int C, int CmaxOut, int H, int W) {
  const int Ho = H >> 1, Wo = W >> 1;
  const int total = 8 * C * Ho * Wo;
  int i = blockIdx.x * blockDim.x + threadIdx.x;
  if (i >= total) return;
  int xo = i % Wo; int t = i / Wo;
  int yo = t % Ho; t /= Ho;
  int c = t % C;   int b = t / C;
  const float* p = in + (((size_t)b * C + c) * H + yo * 2) * W + xo * 2;
  float v = fmaxf(fmaxf(p[0], p[1]), fmaxf(p[W], p[W + 1]));
  out[(((size_t)b * CmaxOut + c) * Ho + yo) * Wo + xo] = v;
}

// ---------------------------------------------------------------------------
// 10) final FC: out[b,c] = flat[b,:] . Wfc[c,:] + bfc[c]  (36864-long dot)
// ---------------------------------------------------------------------------
__global__ void k_fc(const float* __restrict__ flat, const float* __restrict__ Wfc,
                     const float* __restrict__ bfc, float* __restrict__ out) {
  __shared__ float red[256];
  const int c = blockIdx.x, b = blockIdx.y, t = threadIdx.x;
  const float* f = flat + (size_t)b * 36864;
  const float* wv = Wfc + (size_t)c * 36864;
  float s = 0.f;
  for (int i = t; i < 36864; i += 256) s += f[i] * wv[i];
  red[t] = s; __syncthreads();
  for (int k = 128; k > 0; k >>= 1) { if (t < k) red[t] += red[t + k]; __syncthreads(); }
  if (t == 0) out[b * 2 + c] = red[0] + bfc[c];
}

// ---------------------------------------------------------------------------
extern "C" void kernel_launch(void* const* d_in, const int* in_sizes, int n_in,
                              void* d_out, int out_size, void* d_ws, size_t ws_size,
                              hipStream_t stream) {
  (void)in_sizes; (void)out_size; (void)ws_size;
  const float* p_emb  = (const float*)d_in[0];
  const float* h_emb  = (const float*)d_in[1];
  const float* Wt     = (const float*)d_in[2];
  const float* bt     = (const float*)d_in[3];
  const float* w_attn = (const float*)d_in[4];
  const float* Wc     = (const float*)d_in[5];
  const float* bc     = (const float*)d_in[6];
  const float* dw[24]; const float* tw[3]; const float* Wfc; const float* bfc;
  if (n_in >= 36) {            // list inputs expanded into separate tensors
    int i = 7;
    for (int blk = 0; blk < 3; ++blk) {
      for (int l = 0; l < 8; ++l) dw[blk * 8 + l] = (const float*)d_in[i++];
      tw[blk] = (const float*)d_in[i++];
    }
    Wfc = (const float*)d_in[i++]; bfc = (const float*)d_in[i++];
  } else {                     // list inputs concatenated flat
    const int ch0[3] = {38, 99, 129};
    for (int blk = 0; blk < 3; ++blk) {
      const float* base = (const float*)d_in[7 + blk * 2];
      size_t off = 0; int ch = ch0[blk];
      for (int l = 0; l < 8; ++l) { dw[blk * 8 + l] = base + off; off += (size_t)20 * ch * 9; ch += 20; }
      tw[blk] = (const float*)d_in[8 + blk * 2];
    }
    Wfc = (const float*)d_in[13]; bfc = (const float*)d_in[14];
  }

  float* ws  = (float*)d_ws;
  float* xP  = ws;              float* xH  = xP + 131072;
  float* uP  = xH + 131072;     float* mP  = uP + 1024;
  float* uH  = mP + 1024;       float* mH  = uH + 1024;
  float* S   = mH + 1024;
  float* peA = S + 131072;      float* heA = peA + 131072;
  float* actA = heA + 131072;                 // 8*198*128*128 = 25,952,256 f (reused per stage)
  float* bufB = actA + 25952256;              // 8*99*128*128 = 12,976,128 f (transition temp)

  // encode
  k_translate<<<64, 256, 0, stream>>>(p_emb, Wt, bt, xP);
  k_translate<<<64, 256, 0, stream>>>(h_emb, Wt, bt, xH);
  k_rowdot<<<4, 256, 0, stream>>>(xP, w_attn, uP, mP);
  k_rowdot<<<4, 256, 0, stream>>>(xH, w_attn, uH, mH);
  k_scores<<<dim3(8, 8), 256, 0, stream>>>(xP, w_attn, uP, mP, S);
  k_softmax<<<1024, 128, 0, stream>>>(S);
  k_attend<<<dim3(8, 8), 256, 0, stream>>>(S, xP, peA);
  k_scores<<<dim3(8, 8), 256, 0, stream>>>(xH, w_attn, uH, mH, S);
  k_softmax<<<1024, 128, 0, stream>>>(S);
  k_attend<<<dim3(8, 8), 256, 0, stream>>>(S, xH, heA);

  // interaction + input 1x1 conv -> actA channels [0,38), capacity 198
  k_interact<<<dim3(8, 128, 8), 96, 0, stream>>>(peA, heA, Wc, bc, actA, 198);

  // dense block 1 @128x128 (capacity 198), trans1 -> 99ch, pool -> 64x64 (cap 259)
  int Cin = 38;
  for (int l = 0; l < 8; ++l) { k_conv3<<<dim3(128, 8), 256, 0, stream>>>(actA, dw[l], 198, Cin, 128, 128); Cin += 20; }
  { int mt = 7, nt = 1024;
    k_conv1<<<dim3((mt * nt + 7) / 8, 8), 256, 0, stream>>>(actA, 198, 198, tw[0], bufB, 99, 128, 128, mt, nt);
    int total = 8 * 99 * 64 * 64;
    k_pool<<<(total + 255) / 256, 256, 0, stream>>>(bufB, actA, 99, 259, 128, 128); }

  // dense block 2 @64x64 (capacity 259), trans2 -> 129ch, pool -> 32x32 (cap 289)
  Cin = 99;
  for (int l = 0; l < 8; ++l) { k_conv3<<<dim3(32, 8), 256, 0, stream>>>(actA, dw[8 + l], 259, Cin, 64, 64); Cin += 20; }
  { int mt = 9, nt = 256;
    k_conv1<<<dim3((mt * nt + 7) / 8, 8), 256, 0, stream>>>(actA, 259, 259, tw[1], bufB, 129, 64, 64, mt, nt);
    int total = 8 * 129 * 32 * 32;
    k_pool<<<(total + 255) / 256, 256, 0, stream>>>(bufB, actA, 129, 289, 64, 64); }

  // dense block 3 @32x32 (capacity 289), trans3 -> 144ch, pool -> 16x16 (contiguous)
  Cin = 129;
  for (int l = 0; l < 8; ++l) { k_conv3<<<dim3(8, 8), 256, 0, stream>>>(actA, dw[16 + l], 289, Cin, 32, 32); Cin += 20; }
  { int mt = 9, nt = 64;
    k_conv1<<<dim3((mt * nt + 7) / 8, 8), 256, 0, stream>>>(actA, 289, 289, tw[2], bufB, 144, 32, 32, mt, nt);
    int total = 8 * 144 * 16 * 16;
    k_pool<<<(total + 255) / 256, 256, 0, stream>>>(bufB, actA, 144, 144, 32, 32); }

  // classifier
  k_fc<<<dim3(2, 8), 256, 0, stream>>>(actA, Wfc, bfc, (float*)d_out);
}
